// Attention_94489281018
// MI455X (gfx1250) — matrix operand order repaired and checked by reading, not compiled, not run on hardware
//
#include <hip/hip_runtime.h>

typedef __attribute__((ext_vector_type(16))) _Float16 v16h;
typedef __attribute__((ext_vector_type(8)))  float    v8f;
typedef __attribute__((ext_vector_type(2)))  _Float16 h2;
typedef __attribute__((ext_vector_type(2)))  float    f2;

#define B_  4
#define S_  4096
#define D_  512
#define H_  64
#define SCALE_ 0.125f
#define NEGM_  -20.0f

static __device__ __forceinline__ v8f wmma16(v16h a, v16h b, v8f c) {
  // D = A(16x32 f16) * B(32x16 f16) + C(16x16 f32)
  return __builtin_amdgcn_wmma_f32_16x16x32_f16(false, a, false, b, (short)0, c,
                                                false, false);
}

// ---------------------------------------------------------------------------
// Kernel 0: pack Wq/Wk/Wv fp32 [512,64] -> f16 transposed Wt[3][64][512]
// ---------------------------------------------------------------------------
__global__ void wmma_weight_pack(const float* __restrict__ Wq,
                                 const float* __restrict__ Wk,
                                 const float* __restrict__ Wv,
                                 _Float16* __restrict__ Wt) {
  int idx = blockIdx.x * blockDim.x + threadIdx.x;
  if (idx >= 3 * D_ * H_) return;
  int m = idx / (D_ * H_);
  int r = idx % (D_ * H_);
  int d = r / H_;
  int h = r % H_;
  const float* W = (m == 0) ? Wq : (m == 1) ? Wk : Wv;
  Wt[(size_t)m * H_ * D_ + (size_t)h * D_ + d] = (_Float16)W[d * H_ + h];
}

// ---------------------------------------------------------------------------
// Kernel 1: QKV projection with v_wmma_f32_16x16x32_f16.
// grid = B*S/16 blocks of 128 threads; wave w computes h-tile w (16 cols)
// of Q,K,V for a 16-row tile.  V is written transposed: Vt[b][64][S].
// ---------------------------------------------------------------------------
__global__ __launch_bounds__(128) void qkv_proj(
    const float* __restrict__ x, const _Float16* __restrict__ Wt,
    _Float16* __restrict__ Q, _Float16* __restrict__ K,
    _Float16* __restrict__ Vt) {
  const int lane = threadIdx.x & 31;
  const int wave = threadIdx.x >> 5;
  const int hi   = lane >> 4;   // which 16-lane half
  const int ln   = lane & 15;
  const int tile = blockIdx.x;           // 0 .. B*S/16-1
  const int b    = tile >> 8;            // / (S/16)
  const int q0   = (tile & 255) << 4;    // row base within sequence
  const int hb   = wave * 16;            // h-tile base

  const float* xrow = x + ((size_t)b * S_ + q0 + ln) * D_;

  v8f cq = {}; v8f ck = {}; v8f cv = {};

  for (int t = 0; t < D_ / 32; ++t) {
    // A fragment: 16x32 tile of x (fp32 -> f16), contiguous k = 16*hi + s per lane half
    v16h a;
#pragma unroll
    for (int vp = 0; vp < 8; ++vp) {
      int d = t * 32 + hi * 16 + vp * 2;
      f2 xv = *(const f2*)(xrow + d);
      a[2 * vp]     = (_Float16)xv.x;
      a[2 * vp + 1] = (_Float16)xv.y;
    }
    // B fragments: Wt[m][hb+ln][t*32 + hi*16 ..+16] -> contiguous 16 halves
    const int dof = t * 32 + hi * 16;
    const _Float16* wq = Wt + (size_t)(hb + ln) * D_ + dof;
    v16h bq = *(const v16h*)wq;
    v16h bk = *(const v16h*)(wq + (size_t)H_ * D_);
    v16h bv = *(const v16h*)(wq + (size_t)2 * H_ * D_);
    cq = wmma16(a, bq, cq);
    ck = wmma16(a, bk, ck);
    cv = wmma16(a, bv, cv);
  }

#pragma unroll
  for (int i = 0; i < 8; ++i) {
    int row = q0 + i + 8 * hi;                        // C layout: M = i + 8*hi
    size_t o = ((size_t)b * S_ + row) * H_ + hb + ln; // N = ln
    Q[o] = (_Float16)cq[i];
    K[o] = (_Float16)ck[i];
    Vt[(size_t)b * H_ * S_ + (size_t)(hb + ln) * S_ + row] = (_Float16)cv[i];
  }
}

// ---------------------------------------------------------------------------
// Kernel 2: flash attention, additive -20 mask over the FULL row (matches
// reference softmax which includes penalized future keys).
// grid = B * S/64 blocks of 128 threads; each wave owns 16 query rows and
// processes 64 keys per iteration (16 WMMAs / iteration) so the shuffle
// reductions and O-rescale are amortized over twice as many keys.
// ---------------------------------------------------------------------------
__global__ __launch_bounds__(128) void attn_fwd(
    const _Float16* __restrict__ Q, const _Float16* __restrict__ K,
    const _Float16* __restrict__ Vt, float* __restrict__ out) {
  __shared__ _Float16 lds[4][16 * 64];  // per-wave P tile (16 q x 64 k)

  const int lane = threadIdx.x & 31;
  const int wave = threadIdx.x >> 5;
  const int hi   = lane >> 4;
  const int ln   = lane & 15;
  const int b    = blockIdx.x >> 6;               // / (S/64)
  const int q0   = (blockIdx.x & 63) * 64 + wave * 16;

  const _Float16* Qb = Q + (size_t)b * S_ * H_;
  const _Float16* Kb = K + (size_t)b * S_ * H_;
  const _Float16* Vb = Vt + (size_t)b * H_ * S_;

  // Q as two A-fragments (H=64 = 2 K-steps of 32), loaded once
  v16h qa[2];
  const _Float16* qrow = Qb + (size_t)(q0 + ln) * H_;
#pragma unroll
  for (int t = 0; t < 2; ++t) {
#pragma unroll
    for (int vp = 0; vp < 8; ++vp) {
      int h = t * 32 + hi * 16 + vp * 2;
      h2 p = *(const h2*)(qrow + h);
      qa[t][2 * vp]     = p.x;
      qa[t][2 * vp + 1] = p.y;
    }
  }

  v8f o0 = {}, o1 = {}, o2 = {}, o3 = {};
  float m[8], l[8];
#pragma unroll
  for (int i = 0; i < 8; ++i) { m[i] = -1e30f; l[i] = 0.0f; }

  _Float16* pt = &lds[wave][0];

  for (int jb = 0; jb < S_ / 64; ++jb) {
    const int k0 = jb * 64;
    // prefetch next key block (speculative; dropped if OOB)
    __builtin_prefetch(Kb + (size_t)(k0 + 64 + ln) * H_, 0, 1);

    // ---- scores: four 16-key tiles, K-dim = H = 64 (2 wmma steps each) ----
    v8f s[4];
#pragma unroll
    for (int c = 0; c < 4; ++c) s[c] = (v8f){};
#pragma unroll
    for (int t = 0; t < 2; ++t) {
      const _Float16* kr = Kb + (size_t)(k0 + ln) * H_ + t * 32 + hi * 16;
#pragma unroll
      for (int c = 0; c < 4; ++c) {
        v16h bk = *(const v16h*)(kr + (size_t)(c * 16) * H_); // keys k0+c*16+ln
        s[c] = wmma16(qa[t], bk, s[c]);
      }
    }

    // ---- scale + additive causal penalty + online softmax ----
    float pm[8];
#pragma unroll
    for (int i = 0; i < 8; ++i) {
      int qr = q0 + i + 8 * hi;
      float mx = -1e30f;
#pragma unroll
      for (int c = 0; c < 4; ++c) {
        float a = s[c][i] * SCALE_ + ((k0 + c * 16 + ln) > qr ? NEGM_ : 0.0f);
        s[c][i] = a;
        mx = fmaxf(mx, a);
      }
      pm[i] = mx;
    }
#pragma unroll
    for (int off = 8; off >= 1; off >>= 1)
#pragma unroll
      for (int i = 0; i < 8; ++i)
        pm[i] = fmaxf(pm[i], __shfl_xor(pm[i], off, 32));

    float alpha[8], rs[8];
#pragma unroll
    for (int i = 0; i < 8; ++i) {
      float mn = fmaxf(m[i], pm[i]);
      alpha[i] = __expf(m[i] - mn);
      m[i] = mn;
      float acc = 0.0f;
#pragma unroll
      for (int c = 0; c < 4; ++c) {
        float p = __expf(s[c][i] - mn);
        s[c][i] = p;
        acc += p;
      }
      rs[i] = acc;
    }
#pragma unroll
    for (int off = 8; off >= 1; off >>= 1)
#pragma unroll
      for (int i = 0; i < 8; ++i)
        rs[i] += __shfl_xor(rs[i], off, 32);
#pragma unroll
    for (int i = 0; i < 8; ++i) {
      l[i] = l[i] * alpha[i] + rs[i];
      o0[i] *= alpha[i]; o1[i] *= alpha[i];
      o2[i] *= alpha[i]; o3[i] *= alpha[i];
    }

    // ---- relayout P (C-fragments) -> two A-fragments through LDS ----
#pragma unroll
    for (int i = 0; i < 8; ++i) {
      int row = i + 8 * hi;
#pragma unroll
      for (int c = 0; c < 4; ++c)
        pt[row * 64 + c * 16 + ln] = (_Float16)s[c][i];
    }
    v16h pa[2];
#pragma unroll
    for (int f = 0; f < 2; ++f) {
#pragma unroll
      for (int vp = 0; vp < 8; ++vp) {
        int c = f * 32 + hi * 16 + vp * 2;
        h2 p = *(const h2*)(pt + ln * 64 + c);
        pa[f][2 * vp]     = p.x;
        pa[f][2 * vp + 1] = p.y;
      }
    }

    // ---- PV: Vt[b][h][s] gives contiguous B-fragments; 4 h-tiles x 2 ----
    const _Float16* vb = Vb + (size_t)ln * S_ + k0 + hi * 16;
    v8f* oacc[4] = {&o0, &o1, &o2, &o3};
#pragma unroll
    for (int n = 0; n < 4; ++n) {
      const _Float16* vn = vb + (size_t)(n * 16) * S_;
      v16h vf0 = *(const v16h*)(vn);        // keys k0 .. k0+31
      v16h vf1 = *(const v16h*)(vn + 32);   // keys k0+32 .. k0+63
      *oacc[n] = wmma16(pa[0], vf0, *oacc[n]);
      *oacc[n] = wmma16(pa[1], vf1, *oacc[n]);
    }
  }

  // ---- normalize and store fp32 output ----
#pragma unroll
  for (int i = 0; i < 8; ++i) {
    int row = q0 + i + 8 * hi;
    float inv = 1.0f / l[i];
    size_t base = ((size_t)b * S_ + row) * H_;
    out[base +  0 + ln] = o0[i] * inv;
    out[base + 16 + ln] = o1[i] * inv;
    out[base + 32 + ln] = o2[i] * inv;
    out[base + 48 + ln] = o3[i] * inv;
  }
}

// ---------------------------------------------------------------------------
extern "C" void kernel_launch(void* const* d_in, const int* in_sizes, int n_in,
                              void* d_out, int out_size, void* d_ws,
                              size_t ws_size, hipStream_t stream) {
  const float* x  = (const float*)d_in[0];
  const float* Wq = (const float*)d_in[1];
  const float* Wk = (const float*)d_in[2];
  const float* Wv = (const float*)d_in[3];
  float* out = (float*)d_out;

  char* ws = (char*)d_ws;
  const size_t WT_BYTES  = (size_t)3 * H_ * D_ * sizeof(_Float16);  // 196608
  const size_t QKV_BYTES = (size_t)B_ * S_ * H_ * sizeof(_Float16); // 2 MiB
  _Float16* Wt = (_Float16*)ws;
  _Float16* Q  = (_Float16*)(ws + WT_BYTES);
  _Float16* Km = (_Float16*)(ws + WT_BYTES + QKV_BYTES);
  _Float16* Vt = (_Float16*)(ws + WT_BYTES + 2 * QKV_BYTES);

  // 1) pack weights fp32 -> f16 transposed
  hipLaunchKernelGGL(wmma_weight_pack, dim3((3 * D_ * H_ + 255) / 256),
                     dim3(256), 0, stream, Wq, Wk, Wv, Wt);
  // 2) QKV projection (WMMA)
  hipLaunchKernelGGL(qkv_proj, dim3(B_ * S_ / 16), dim3(128), 0, stream,
                     x, Wt, Q, Km, Vt);
  // 3) flash attention with additive -20 causal penalty (WMMA)
  hipLaunchKernelGGL(attn_fwd, dim3(B_ * (S_ / 64)), dim3(128), 0, stream,
                     Q, Km, Vt, out);
}
